// BasicTransformerBlock_17085379903983
// MI455X (gfx1250) — matrix operand order repaired and checked by reading, not compiled, not run on hardware
//
#include <hip/hip_runtime.h>
#include <hip/hip_bf16.h>

// ---------------------------------------------------------------------------
// MI455X / gfx1250: bf16 WMMA (V_WMMA_F32_16X16X32_BF16) GEMMs + single-wave
// flash attention, with GLOBAL_LOAD_ASYNC_TO_LDS_B128 tile staging (ASYNCcnt).
// All bf16 data handled as raw unsigned short (bit patterns).
// ---------------------------------------------------------------------------

typedef __attribute__((ext_vector_type(16))) __bf16 v16bf;
typedef __attribute__((ext_vector_type(8)))  float  v8f;

// Sizes from the reference
#define BB   4
#define NN   2048
#define DD   512
#define CNN  77
#define CDD  768
#define HH   8
#define HDD  64
#define II   512           // H*HD
#define FFF  2048          // FF (only first quarter of ff1 output is used)
#define MTOK 8192          // B*N rows

__device__ __forceinline__ unsigned short f32_to_bf16(float f) {
  unsigned int u = __builtin_bit_cast(unsigned int, f);
  unsigned int r = u + 0x7fffu + ((u >> 16) & 1u);   // round-to-nearest-even
  return (unsigned short)(r >> 16);
}

union Frag { uint4 q[2]; v16bf v; };

// Async DMA: global -> LDS, 16 bytes per lane. lds_off is the per-lane LDS
// byte offset (low 32 bits of a generic pointer to __shared__), gaddr is the
// per-lane global address. Tracked by ASYNCcnt.
__device__ __forceinline__ void async_copy_b128(unsigned lds_off, const void* gaddr) {
  asm volatile("global_load_async_to_lds_b128 %0, %1, off"
               :: "v"(lds_off), "v"((unsigned long long)(size_t)gaddr)
               : "memory");
}

__device__ __forceinline__ void wait_async0() {
#if __has_builtin(__builtin_amdgcn_s_wait_asynccnt)
  __builtin_amdgcn_s_wait_asynccnt(0);
#else
  asm volatile("s_wait_asynccnt 0x0" ::: "memory");
#endif
}

// ---------------------------------------------------------------------------
// fp32 -> bf16 elementwise convert
// ---------------------------------------------------------------------------
__global__ void k_f32_to_bf16(const float* __restrict__ in,
                              unsigned short* __restrict__ out, int n) {
  int i = blockIdx.x * 256 + threadIdx.x;
  if (i < n) out[i] = f32_to_bf16(in[i]);
}

// ---------------------------------------------------------------------------
// LayerNorm over D=512, one wave per row, bf16 output.  block = 256 (8 rows)
// ---------------------------------------------------------------------------
__global__ void k_layernorm(const float* __restrict__ x,
                            const float* __restrict__ g,
                            const float* __restrict__ b,
                            unsigned short* __restrict__ o, int M) {
  int row  = blockIdx.x * 8 + (threadIdx.x >> 5);
  int lane = threadIdx.x & 31;
  if (row >= M) return;
  const float4* xp = (const float4*)(x + (size_t)row * DD);
  float4 v[4];
  float s = 0.f, ss = 0.f;
#pragma unroll
  for (int i = 0; i < 4; ++i) {
    v[i] = xp[lane + 32 * i];
    s  += v[i].x + v[i].y + v[i].z + v[i].w;
    ss += v[i].x * v[i].x + v[i].y * v[i].y + v[i].z * v[i].z + v[i].w * v[i].w;
  }
#pragma unroll
  for (int off = 16; off; off >>= 1) {
    s  += __shfl_xor(s,  off);
    ss += __shfl_xor(ss, off);
  }
  float mu  = s * (1.f / DD);
  float var = ss * (1.f / DD) - mu * mu;
  float rs  = rsqrtf(var + 1e-5f);
#pragma unroll
  for (int i = 0; i < 4; ++i) {
    int c = (lane + 32 * i) * 4;
    float4 gg = *(const float4*)(g + c);
    float4 bb = *(const float4*)(b + c);
    unsigned int h0 = f32_to_bf16((v[i].x - mu) * rs * gg.x + bb.x);
    unsigned int h1 = f32_to_bf16((v[i].y - mu) * rs * gg.y + bb.y);
    unsigned int h2 = f32_to_bf16((v[i].z - mu) * rs * gg.z + bb.z);
    unsigned int h3 = f32_to_bf16((v[i].w - mu) * rs * gg.w + bb.w);
    uint2 u; u.x = h0 | (h1 << 16); u.y = h2 | (h3 << 16);
    *(uint2*)(o + (size_t)row * DD + c) = u;
  }
}

// ---------------------------------------------------------------------------
// bf16 GEMM: C[M,N] = A[M,K] * W[K,N] (+bias) (+fp32 residual)
// block = 128 threads (4 waves); 64x64 tile; each wave 32x32 (2x2 WMMA tiles)
// A tile staged via async DMA to LDS; B tile staged transposed (manual).
// Requires K%32==0, N%64==0; M arbitrary (guarded; stale LDS rows >= M only
// influence unstored C rows).
// ---------------------------------------------------------------------------
__global__ void k_gemm_bf16(const unsigned short* __restrict__ A, int lda,
                            const unsigned short* __restrict__ W, int ldw,
                            const float* __restrict__ bias,
                            const float* __restrict__ res,
                            unsigned short* __restrict__ outb,
                            float* __restrict__ outf,
                            int M, int N, int K) {
  __shared__ unsigned short As[64][40];   // [m][k]   (+8 pad)
  __shared__ unsigned short Bs[64][40];   // [n][k]   transposed (+8 pad)

  const int tid  = threadIdx.x;
  const int lane = tid & 31, l15 = lane & 15, hi = lane >> 4;
  const int wave = tid >> 5;
  const int wm   = wave >> 1, wn = wave & 1;
  const int m0   = blockIdx.y * 64, n0 = blockIdx.x * 64;

  v8f acc[2][2] = {{{}, {}}, {{}, {}}};

  const int ar = tid >> 1,  ac = (tid & 1) * 16;   // A staging coords
  const int kr = tid >> 2,  nc = (tid & 3) * 16;   // B staging coords
  const int gr = m0 + ar;
  const unsigned a_lds = (unsigned)(size_t)&As[ar][ac];

  for (int k0 = 0; k0 < K; k0 += 32) {
    // ---- stage A tile (64x32) via async DMA ----
    if (gr < M) {
      const unsigned short* ap = A + (size_t)gr * lda + k0 + ac;
      async_copy_b128(a_lds,      ap);
      async_copy_b128(a_lds + 16, ap + 8);
      if (k0 + 32 < K)
        __builtin_prefetch(ap + 32, 0, 3);
    }
    // ---- stage B tile (32x64) transposed into Bs[n][k] ----
    {
      const uint4* p = (const uint4*)(W + (size_t)(k0 + kr) * ldw + n0 + nc);
      union { uint4 q[2]; unsigned short h[16]; } u;
      u.q[0] = p[0]; u.q[1] = p[1];
      if (k0 + 32 < K)
        __builtin_prefetch(W + (size_t)(k0 + 32 + kr) * ldw + n0 + nc, 0, 3);
#pragma unroll
      for (int i = 0; i < 16; ++i) Bs[nc + i][kr] = u.h[i];
    }
    wait_async0();
    __syncthreads();

    // ---- fragments + WMMA ----
    Frag a[2], bf[2];
#pragma unroll
    for (int mi = 0; mi < 2; ++mi) {
      int r = wm * 32 + mi * 16 + l15;
      a[mi].q[0] = *(const uint4*)&As[r][hi * 16];       // K = hi*16 + 0..7
      a[mi].q[1] = *(const uint4*)&As[r][hi * 16 + 8];   // K = hi*16 + 8..15
    }
#pragma unroll
    for (int ni = 0; ni < 2; ++ni) {
      int c = wn * 32 + ni * 16 + l15;
      const uint4* p = (const uint4*)&Bs[c][hi * 16];    // K = hi*16 + 0..15
      bf[ni].q[0] = p[0]; bf[ni].q[1] = p[1];
    }
#pragma unroll
    for (int mi = 0; mi < 2; ++mi)
#pragma unroll
      for (int ni = 0; ni < 2; ++ni)
        acc[mi][ni] = __builtin_amdgcn_wmma_f32_16x16x32_bf16(
            false, a[mi].v, false, bf[ni].v, (short)0, acc[mi][ni], false, false);
    __syncthreads();
  }

  // ---- epilogue ----
#pragma unroll
  for (int mi = 0; mi < 2; ++mi)
#pragma unroll
    for (int ni = 0; ni < 2; ++ni) {
      int gc = n0 + wn * 32 + ni * 16 + l15;
      float bv = bias ? bias[gc] : 0.f;
#pragma unroll
      for (int r = 0; r < 8; ++r) {
        int grow = m0 + wm * 32 + mi * 16 + hi * 8 + r;
        if (grow < M) {
          float v = acc[mi][ni][r] + bv;
          if (res)  v += res[(size_t)grow * N + gc];
          if (outf) outf[(size_t)grow * N + gc] = v;
          else      outb[(size_t)grow * N + gc] = f32_to_bf16(v);
        }
      }
    }
}

// ---------------------------------------------------------------------------
// Flash attention: 1 wave per (batch, head, 16-query tile).
// Q rows: 2048/batch.  K/V rows: nkv/batch (2048 self, 77 cross).
// K/V tiles (32 keys x 64 hd) staged to LDS by async DMA (1 row per lane).
// ---------------------------------------------------------------------------
__global__ void k_flash_attn(const unsigned short* __restrict__ Q,
                             const unsigned short* __restrict__ Kp,
                             const unsigned short* __restrict__ Vp,
                             unsigned short* __restrict__ O, int nkv) {
  __shared__ unsigned short Ks[32][72];   // [key][hd] (+8 pad)
  __shared__ unsigned short Vs[32][72];   // [key][hd] (+8 pad)
  __shared__ unsigned short P[16][40];    // softmax probs, C->A relayout

  const int NQT = NN / 16;
  int bid = blockIdx.x;
  int qt  = bid % NQT;
  int h   = (bid / NQT) % HH;
  int b   = bid / (NQT * HH);
  int lane = threadIdx.x & 31, l15 = lane & 15, hi = lane >> 4;

  size_t qrow0  = (size_t)b * NN + (size_t)qt * 16;
  size_t kvrow0 = (size_t)b * nkv;

  // Q fragments (A-layout, 16x32 each, covering hd 0..31 / 32..63)
  const unsigned short* qbase = Q + (qrow0 + l15) * II + h * HDD;
  Frag qf[2];
#pragma unroll
  for (int f = 0; f < 2; ++f) {
    qf[f].q[0] = *(const uint4*)(qbase + f * 32 + hi * 16);
    qf[f].q[1] = *(const uint4*)(qbase + f * 32 + hi * 16 + 8);
  }

  v8f ot[4] = {{}, {}, {}, {}};
  float mrow[8], lrow[8];
#pragma unroll
  for (int r = 0; r < 8; ++r) { mrow[r] = -3.0e38f; lrow[r] = 0.f; }

  const unsigned k_lds = (unsigned)(size_t)&Ks[lane][0];
  const unsigned v_lds = (unsigned)(size_t)&Vs[lane][0];

  int kend = ((nkv + 31) / 32) * 32;
  for (int kb = 0; kb < kend; kb += 32) {
    // ---- stage K/V rows (128B each) via async DMA; zero OOB V rows ----
    {
      int key = kb + lane;
      if (key < nkv) {
        const unsigned short* kp = Kp + (kvrow0 + key) * II + h * HDD;
        const unsigned short* vp = Vp + (kvrow0 + key) * II + h * HDD;
#pragma unroll
        for (int c = 0; c < 8; ++c) {
          async_copy_b128(k_lds + c * 16, kp + c * 8);
          async_copy_b128(v_lds + c * 16, vp + c * 8);
        }
      } else {
        uint4 z{};
        uint4* vz = (uint4*)&Vs[lane][0];
#pragma unroll
        for (int c = 0; c < 8; ++c) vz[c] = z;   // 0 * stale = NaN, so zero V
      }
      wait_async0();
      __syncthreads();
    }

    // ---- scores: S = Q K^T * SCALE, masked beyond nkv ----
    v8f s[2];
#pragma unroll
    for (int t = 0; t < 2; ++t) {
      const unsigned short* krow = &Ks[t * 16 + l15][0];
      Frag kf0, kf1;
      kf0.q[0] = *(const uint4*)(krow + hi * 16);        // hd = hi*16 + 0..7
      kf0.q[1] = *(const uint4*)(krow + hi * 16 + 8);    // hd = hi*16 + 8..15
      kf1.q[0] = *(const uint4*)(krow + 32 + hi * 16);
      kf1.q[1] = *(const uint4*)(krow + 32 + hi * 16 + 8);
      v8f a = {};
      a = __builtin_amdgcn_wmma_f32_16x16x32_bf16(false, qf[0].v, false, kf0.v,
                                                  (short)0, a, false, false);
      a = __builtin_amdgcn_wmma_f32_16x16x32_bf16(false, qf[1].v, false, kf1.v,
                                                  (short)0, a, false, false);
      bool oob = (kb + t * 16 + l15) >= nkv;
#pragma unroll
      for (int r = 0; r < 8; ++r)
        s[t][r] = oob ? -3.0e38f : a[r] * 0.125f;        // SCALE = 64^-0.5
    }

    // ---- online softmax (reduce over 16 lanes within each half-group) ----
    float rm[8];
#pragma unroll
    for (int r = 0; r < 8; ++r) rm[r] = fmaxf(s[0][r], s[1][r]);
#pragma unroll
    for (int off = 1; off < 16; off <<= 1)
#pragma unroll
      for (int r = 0; r < 8; ++r) rm[r] = fmaxf(rm[r], __shfl_xor(rm[r], off));

    float alpha[8];
#pragma unroll
    for (int r = 0; r < 8; ++r) {
      float mn = fmaxf(mrow[r], rm[r]);
      alpha[r] = __expf(mrow[r] - mn);
      mrow[r]  = mn;
    }

    v8f p0, p1;
    float rs[8];
#pragma unroll
    for (int r = 0; r < 8; ++r) {
      p0[r] = __expf(s[0][r] - mrow[r]);
      p1[r] = __expf(s[1][r] - mrow[r]);
      rs[r] = p0[r] + p1[r];
    }
#pragma unroll
    for (int off = 1; off < 16; off <<= 1)
#pragma unroll
      for (int r = 0; r < 8; ++r) rs[r] += __shfl_xor(rs[r], off);
#pragma unroll
    for (int r = 0; r < 8; ++r) {
      lrow[r] = lrow[r] * alpha[r] + rs[r];
      ot[0][r] *= alpha[r]; ot[1][r] *= alpha[r];
      ot[2][r] *= alpha[r]; ot[3][r] *= alpha[r];
    }

    // ---- C-layout -> A-layout transpose of P through LDS ----
    __syncthreads();
#pragma unroll
    for (int r = 0; r < 8; ++r) {
      P[hi * 8 + r][l15]      = f32_to_bf16(p0[r]);
      P[hi * 8 + r][16 + l15] = f32_to_bf16(p1[r]);
    }
    __syncthreads();
    Frag pf;
    pf.q[0] = *(const uint4*)&P[l15][hi * 16];
    pf.q[1] = *(const uint4*)&P[l15][hi * 16 + 8];

    // ---- O += P * V  (4 output hd-tiles) ----
#pragma unroll
    for (int t2 = 0; t2 < 4; ++t2) {
      union { unsigned short hh[16]; v16bf v; } vf;
#pragma unroll
      for (int e = 0; e < 16; ++e)
        vf.hh[e] = Vs[hi * 16 + e][t2 * 16 + l15];
      ot[t2] = __builtin_amdgcn_wmma_f32_16x16x32_bf16(
          false, pf.v, false, vf.v, (short)0, ot[t2], false, false);
    }
    __syncthreads();
  }

  // ---- normalize + store bf16 ----
#pragma unroll
  for (int r = 0; r < 8; ++r) {
    float inv = 1.0f / lrow[r];
    size_t row = qrow0 + hi * 8 + r;
    unsigned short* op = O + row * II + h * HDD + l15;
    op[0]  = f32_to_bf16(ot[0][r] * inv);
    op[16] = f32_to_bf16(ot[1][r] * inv);
    op[32] = f32_to_bf16(ot[2][r] * inv);
    op[48] = f32_to_bf16(ot[3][r] * inv);
  }
}

// ---------------------------------------------------------------------------
// Launch
// ---------------------------------------------------------------------------
extern "C" void kernel_launch(void* const* d_in, const int* in_sizes, int n_in,
                              void* d_out, int out_size, void* d_ws, size_t ws_size,
                              hipStream_t stream) {
  const float* x     = (const float*)d_in[0];
  const float* ctx   = (const float*)d_in[1];
  const float* ln1_g = (const float*)d_in[2];
  const float* ln1_b = (const float*)d_in[3];
  const float* ln2_g = (const float*)d_in[4];
  const float* ln2_b = (const float*)d_in[5];
  const float* ln3_g = (const float*)d_in[6];
  const float* ln3_b = (const float*)d_in[7];
  const float* q1_w  = (const float*)d_in[8];
  const float* k1_w  = (const float*)d_in[9];
  const float* v1_w  = (const float*)d_in[10];
  const float* o1_w  = (const float*)d_in[11];
  const float* o1_b  = (const float*)d_in[12];
  const float* q2_w  = (const float*)d_in[13];
  const float* k2_w  = (const float*)d_in[14];
  const float* v2_w  = (const float*)d_in[15];
  const float* o2_w  = (const float*)d_in[16];
  const float* o2_b  = (const float*)d_in[17];
  const float* ff1_w = (const float*)d_in[18];
  const float* ff1_b = (const float*)d_in[19];
  const float* ff2_w = (const float*)d_in[20];
  const float* ff2_b = (const float*)d_in[21];
  float* out = (float*)d_out;

  // scratch carve-out (~125 MB total)
  char* wp = (char*)d_ws;
  auto carve = [&](size_t bytes) {
    char* p = wp;
    wp += (bytes + 255) & ~(size_t)255;
    return p;
  };
  unsigned short* wq1  = (unsigned short*)carve((size_t)DD * II * 2);
  unsigned short* wk1  = (unsigned short*)carve((size_t)DD * II * 2);
  unsigned short* wv1  = (unsigned short*)carve((size_t)DD * II * 2);
  unsigned short* wo1  = (unsigned short*)carve((size_t)II * DD * 2);
  unsigned short* wq2  = (unsigned short*)carve((size_t)DD * II * 2);
  unsigned short* wk2  = (unsigned short*)carve((size_t)CDD * II * 2);
  unsigned short* wv2  = (unsigned short*)carve((size_t)CDD * II * 2);
  unsigned short* wo2  = (unsigned short*)carve((size_t)II * DD * 2);
  unsigned short* wff1 = (unsigned short*)carve((size_t)DD * 8192 * 2);
  unsigned short* wff2 = (unsigned short*)carve((size_t)FFF * DD * 2);
  unsigned short* ctxb = (unsigned short*)carve((size_t)BB * CNN * CDD * 2);
  unsigned short* xnb  = (unsigned short*)carve((size_t)MTOK * DD * 2);
  unsigned short* qb   = (unsigned short*)carve((size_t)MTOK * II * 2);
  unsigned short* kb   = (unsigned short*)carve((size_t)MTOK * II * 2);
  unsigned short* vb   = (unsigned short*)carve((size_t)MTOK * II * 2);
  unsigned short* atb  = (unsigned short*)carve((size_t)MTOK * II * 2);
  float*          x1f  = (float*)carve((size_t)MTOK * DD * 4);
  float*          x2f  = (float*)carve((size_t)MTOK * DD * 4);
  unsigned short* hb   = (unsigned short*)carve((size_t)MTOK * FFF * 2);

  auto conv = [&](const float* src, unsigned short* dst, int n) {
    k_f32_to_bf16<<<(n + 255) / 256, 256, 0, stream>>>(src, dst, n);
  };
  conv(q1_w, wq1, DD * II);
  conv(k1_w, wk1, DD * II);
  conv(v1_w, wv1, DD * II);
  conv(o1_w, wo1, II * DD);
  conv(q2_w, wq2, DD * II);
  conv(k2_w, wk2, CDD * II);
  conv(v2_w, wv2, CDD * II);
  conv(o2_w, wo2, II * DD);
  conv(ff1_w, wff1, DD * 8192);
  conv(ff2_w, wff2, FFF * DD);
  conv(ctx, ctxb, BB * CNN * CDD);

  auto gemm = [&](const unsigned short* A, int lda, const unsigned short* W,
                  int ldw, const float* bias, const float* res,
                  unsigned short* ob, float* of, int M, int N, int K) {
    dim3 grid(N / 64, (M + 63) / 64);
    k_gemm_bf16<<<grid, 128, 0, stream>>>(A, lda, W, ldw, bias, res, ob, of,
                                          M, N, K);
  };

  const int FLASH_BLOCKS = BB * HH * (NN / 16);

  // ---- block 1: self attention ----
  k_layernorm<<<MTOK / 8, 256, 0, stream>>>(x, ln1_g, ln1_b, xnb, MTOK);
  gemm(xnb, DD, wq1, II, nullptr, nullptr, qb, nullptr, MTOK, II, DD);
  gemm(xnb, DD, wk1, II, nullptr, nullptr, kb, nullptr, MTOK, II, DD);
  gemm(xnb, DD, wv1, II, nullptr, nullptr, vb, nullptr, MTOK, II, DD);
  k_flash_attn<<<FLASH_BLOCKS, 32, 0, stream>>>(qb, kb, vb, atb, NN);
  gemm(atb, II, wo1, DD, o1_b, x, nullptr, x1f, MTOK, DD, II);

  // ---- block 2: cross attention ----
  k_layernorm<<<MTOK / 8, 256, 0, stream>>>(x1f, ln2_g, ln2_b, xnb, MTOK);
  gemm(xnb, DD, wq2, II, nullptr, nullptr, qb, nullptr, MTOK, II, DD);
  gemm(ctxb, CDD, wk2, II, nullptr, nullptr, kb, nullptr, BB * CNN, II, CDD);
  gemm(ctxb, CDD, wv2, II, nullptr, nullptr, vb, nullptr, BB * CNN, II, CDD);
  k_flash_attn<<<FLASH_BLOCKS, 32, 0, stream>>>(qb, kb, vb, atb, CNN);
  gemm(atb, II, wo2, DD, o2_b, x1f, nullptr, x2f, MTOK, DD, II);

  // ---- feed forward (only first FF columns of ff1 are used) ----
  k_layernorm<<<MTOK / 8, 256, 0, stream>>>(x2f, ln3_g, ln3_b, xnb, MTOK);
  gemm(xnb, DD, wff1, 8192, ff1_b, nullptr, hb, nullptr, MTOK, FFF, DD);
  gemm(hb, FFF, wff2, DD, ff2_b, x2f, nullptr, out, MTOK, DD, FFF);
}
